// PositionAttentionModule_73332271612436
// MI455X (gfx1250) — matrix operand order repaired and checked by reading, not compiled, not run on hardware
//
#include <hip/hip_runtime.h>

// ---------------------------------------------------------------------------
// Position Attention Module (DANet PAM) for gfx1250 / MI455X
//   B=4, C=512, CR=64, N=64*64=4096
//   GEMMs via v_wmma_f32_16x16x32_f16 (f16 in, f32 accumulate).
//   Dominant GEMM uses double-buffered LDS staging with
//   global_load_async_to_lds_b128 + s_wait_asynccnt (CDNA5 async path).
// ---------------------------------------------------------------------------

typedef __attribute__((ext_vector_type(8)))  _Float16 v8h;
typedef __attribute__((ext_vector_type(16))) _Float16 v16h;
typedef __attribute__((ext_vector_type(8)))  float    v8f;

constexpr int B_   = 4;
constexpr int C_   = 512;
constexpr int CR_  = 64;
constexpr int NPIX = 4096;       // 64*64
constexpr int MALL = 640;        // CR + CR + C  (fused projection rows)

// ---------------- workspace layout (bytes, 256-aligned regions) ------------
constexpr size_t OFF_XT   = 0;                                   // f16 [B][N][C]
constexpr size_t SZ_XT    = (size_t)B_ * NPIX * C_ * 2;          // 16 MiB
constexpr size_t OFF_WALL = OFF_XT + SZ_XT;                      // f16 [640][512]
constexpr size_t SZ_WALL  = (size_t)MALL * C_ * 2;
constexpr size_t OFF_BIAS = OFF_WALL + SZ_WALL;                  // f32 [640]
constexpr size_t SZ_BIAS  = (size_t)MALL * 4;
constexpr size_t OFF_BT   = OFF_BIAS + SZ_BIAS;                  // f16 [B][N][CR]
constexpr size_t SZ_BT    = (size_t)B_ * NPIX * CR_ * 2;
constexpr size_t OFF_CT   = OFF_BT + SZ_BT;                      // f16 [B][N][CR]
constexpr size_t OFF_DF   = OFF_CT + SZ_BT;                      // f16 [B][C][N]
constexpr size_t SZ_DF    = (size_t)B_ * C_ * NPIX * 2;
constexpr size_t OFF_ST   = OFF_DF + SZ_DF;                      // f16 [B][N(j)][N(i)]

// ---------------- WMMA fragment loaders ------------------------------------
// A fragment: 16x32 (MxK), row-major, ptr at tile origin (row0, k0).
// lane L: m = L&15 ; group g = L>>4 : halves h -> K = 16g + h
__device__ __forceinline__ v16h load_frag_a(const _Float16* a, int lda, int lane) {
    const _Float16* p = a + (size_t)(lane & 15) * lda + ((lane >> 4) << 4);
    v8h lo = *(const v8h*)(p);
    v8h hi = *(const v8h*)(p + 8);
    v16h r;
#pragma unroll
    for (int i = 0; i < 8; ++i) { r[i] = lo[i]; r[i + 8] = hi[i]; }
    return r;
}

// B fragment: 32x16 (KxN), column-major (K contiguous per column), ptr at
// (k0, col0), ldb = halves per column.
// lane L: n = L&15 ; halves h -> K = 16*(L>>4) + h  (one 32B contiguous load)
__device__ __forceinline__ v16h load_frag_b(const _Float16* b, int ldb, int lane) {
    const _Float16* p = b + (size_t)(lane & 15) * ldb + ((lane >> 4) << 4);
    return *(const v16h*)p;
}

__device__ __forceinline__ v8f wmma_f16(v16h a, v16h b, v8f c) {
    return __builtin_amdgcn_wmma_f32_16x16x32_f16(false, a, false, b,
                                                  (short)0, c, false, false);
}

__device__ __forceinline__ v8h cvt_v8h(v8f a) {
    v8h r;
#pragma unroll
    for (int i = 0; i < 8; ++i) r[i] = (_Float16)a[i];
    return r;
}

// Per-lane async copy: 16 bytes global -> LDS, tracked by ASYNCcnt.
__device__ __forceinline__ void async_ld16(unsigned lds_byte_off, const _Float16* gptr) {
    asm volatile("global_load_async_to_lds_b128 %0, %1, off"
                 :: "v"(lds_byte_off), "v"(gptr)
                 : "memory");
}

// ---------------- kernel 1: pack weights + bias -----------------------------
__global__ void cast_weights(const float* __restrict__ Wb, const float* __restrict__ bb,
                             const float* __restrict__ Wc, const float* __restrict__ bc,
                             const float* __restrict__ Wd, const float* __restrict__ bd,
                             _Float16* __restrict__ Wall, float* __restrict__ biasAll) {
    int idx = blockIdx.x * 256 + threadIdx.x;       // 0 .. 640*512-1
    int row = idx >> 9;
    int col = idx & 511;
    float w;
    if (row < CR_)            w = Wb[row * C_ + col];
    else if (row < 2 * CR_)   w = Wc[(row - CR_) * C_ + col];
    else                      w = Wd[(row - 2 * CR_) * C_ + col];
    Wall[idx] = (_Float16)w;
    if (idx < MALL) {
        float bv = (idx < CR_) ? bb[idx]
                 : (idx < 2 * CR_) ? bc[idx - CR_]
                 : bd[idx - 2 * CR_];
        biasAll[idx] = bv;
    }
}

// ---------------- kernel 2: x -> xT f16 (transpose via LDS) ----------------
// grid: (N/32, C/32, B), block: (32, 8)
__global__ void cast_transpose_x(const float* __restrict__ x, _Float16* __restrict__ xT) {
    __shared__ float tile[32][33];
    int b  = blockIdx.z;
    int n0 = blockIdx.x * 32;
    int c0 = blockIdx.y * 32;
    int tx = threadIdx.x, ty = threadIdx.y;
    const float* xb = x + (size_t)b * C_ * NPIX;
#pragma unroll
    for (int k = 0; k < 32; k += 8)
        tile[ty + k][tx] = xb[(size_t)(c0 + ty + k) * NPIX + n0 + tx];
    __syncthreads();
    _Float16* xTb = xT + (size_t)b * NPIX * C_;
#pragma unroll
    for (int k = 0; k < 32; k += 8)
        xTb[(size_t)(n0 + ty + k) * C_ + c0 + tx] = (_Float16)tile[tx][ty + k];
}

// ---------------- kernel 3: fused projections (b,c,d) ----------------------
// out[m,n] = sum_k Wall[m,k] * xT[n,k] + bias[m]
// grid: (N/128, MALL/128=5, B), block 128 (4 waves, 2x2)
__global__ __launch_bounds__(128)
void proj_gemm(const _Float16* __restrict__ Wall, const float* __restrict__ biasAll,
               const _Float16* __restrict__ xT,
               _Float16* __restrict__ bT, _Float16* __restrict__ cT,
               _Float16* __restrict__ dF) {
    int lane = threadIdx.x & 31;
    int wave = threadIdx.x >> 5;
    int wm = wave >> 1, wn = wave & 1;
    int batch = blockIdx.z;
    int m0 = blockIdx.y * 128 + wm * 64;
    int n0 = blockIdx.x * 128 + wn * 64;

    const _Float16* A  = Wall + (size_t)m0 * C_;
    const _Float16* Bm = xT + (size_t)batch * NPIX * C_ + (size_t)n0 * C_;

    v8f acc[4][4];
#pragma unroll
    for (int i = 0; i < 4; ++i)
#pragma unroll
        for (int j = 0; j < 4; ++j) acc[i][j] = (v8f)0.f;

    for (int k0 = 0; k0 < C_; k0 += 32) {
        v16h af[4], bf[4];
#pragma unroll
        for (int i = 0; i < 4; ++i) af[i] = load_frag_a(A + (size_t)(16 * i) * C_ + k0, C_, lane);
#pragma unroll
        for (int j = 0; j < 4; ++j) bf[j] = load_frag_b(Bm + (size_t)(16 * j) * C_ + k0, C_, lane);
#pragma unroll
        for (int i = 0; i < 4; ++i)
#pragma unroll
            for (int j = 0; j < 4; ++j) acc[i][j] = wmma_f16(af[i], bf[j], acc[i][j]);
    }

    int lg = lane >> 4, ln = lane & 15;
#pragma unroll
    for (int ti = 0; ti < 4; ++ti) {
        int mtile = m0 + 16 * ti;         // 16-row tile, fully inside one region
        int mbase = mtile + 8 * lg;       // first of 8 contiguous couts in regs
#pragma unroll
        for (int tj = 0; tj < 4; ++tj) {
            int nb = n0 + 16 * tj + ln;
            if (mtile < CR_) {            // -> bT[batch][nb][cout], contiguous
                v8h o;
#pragma unroll
                for (int r = 0; r < 8; ++r)
                    o[r] = (_Float16)(acc[ti][tj][r] + biasAll[mbase + r]);
                *(v8h*)(bT + ((size_t)batch * NPIX + nb) * CR_ + mbase) = o;
            } else if (mtile < 2 * CR_) { // -> cT
                v8h o;
#pragma unroll
                for (int r = 0; r < 8; ++r)
                    o[r] = (_Float16)(acc[ti][tj][r] + biasAll[mbase + r]);
                *(v8h*)(cT + ((size_t)batch * NPIX + nb) * CR_ + (mbase - CR_)) = o;
            } else {                      // -> dF[batch][cout][nb] (row-major)
#pragma unroll
                for (int r = 0; r < 8; ++r) {
                    int cc = mbase - 2 * CR_ + r;
                    dF[((size_t)batch * C_ + cc) * NPIX + nb] =
                        (_Float16)(acc[ti][tj][r] + biasAll[mbase + r]);
                }
            }
        }
    }
}

// ---------------- kernel 4: attention logits s = b^T c ---------------------
// s[i,j] = sum_k bT[i][k] * cT[j][k]; stored transposed: sT[j][i] (f16)
// grid: (N/128 [j], N/128 [i], B), block 128 (2x2 waves)
__global__ __launch_bounds__(128)
void logits_gemm(const _Float16* __restrict__ bT, const _Float16* __restrict__ cT,
                 _Float16* __restrict__ sT) {
    int lane = threadIdx.x & 31;
    int wave = threadIdx.x >> 5;
    int wm = wave >> 1, wn = wave & 1;
    int batch = blockIdx.z;
    int i0 = blockIdx.y * 128 + wm * 64;
    int j0 = blockIdx.x * 128 + wn * 64;

    const _Float16* A  = bT + (size_t)batch * NPIX * CR_ + (size_t)i0 * CR_;
    const _Float16* Bm = cT + (size_t)batch * NPIX * CR_ + (size_t)j0 * CR_;

    v8f acc[4][4];
#pragma unroll
    for (int i = 0; i < 4; ++i)
#pragma unroll
        for (int j = 0; j < 4; ++j) acc[i][j] = (v8f)0.f;

#pragma unroll
    for (int k0 = 0; k0 < CR_; k0 += 32) {
        v16h af[4], bf[4];
#pragma unroll
        for (int i = 0; i < 4; ++i) af[i] = load_frag_a(A + (size_t)(16 * i) * CR_ + k0, CR_, lane);
#pragma unroll
        for (int j = 0; j < 4; ++j) bf[j] = load_frag_b(Bm + (size_t)(16 * j) * CR_ + k0, CR_, lane);
#pragma unroll
        for (int i = 0; i < 4; ++i)
#pragma unroll
            for (int j = 0; j < 4; ++j) acc[i][j] = wmma_f16(af[i], bf[j], acc[i][j]);
    }

    int lg = lane >> 4, ln = lane & 15;
#pragma unroll
    for (int ti = 0; ti < 4; ++ti) {
        int ibase = i0 + 16 * ti + 8 * lg;        // 8 contiguous i per lane
#pragma unroll
        for (int tj = 0; tj < 4; ++tj) {
            int j = j0 + 16 * tj + ln;
            *(v8h*)(sT + ((size_t)batch * NPIX + j) * NPIX + ibase) = cvt_v8h(acc[ti][tj]);
        }
    }
}

// ---------------- kernel 5: softmax over j (in place on sT) ----------------
// sT layout [j][i]; softmax for fixed i over j == down a column of sT.
// grid: (N/256, B), block 256
__global__ void softmax_inplace(_Float16* __restrict__ sT) {
    int batch = blockIdx.y;
    int i = blockIdx.x * 256 + threadIdx.x;
    _Float16* base = sT + (size_t)batch * NPIX * NPIX + i;
    float m = -__builtin_inff(), s = 0.f;
    for (int j = 0; j < NPIX; ++j) {
        float v = (float)base[(size_t)j * NPIX];
        float nm = fmaxf(m, v);
        s = s * __expf(m - nm) + __expf(v - nm);
        m = nm;
    }
    float inv = 1.f / s;
    for (int j = 0; j < NPIX; ++j) {
        float v = (float)base[(size_t)j * NPIX];
        base[(size_t)j * NPIX] = (_Float16)(__expf(v - m) * inv);
    }
}

// ---------------- kernel 6: y^T = sT * d^T, fused residual -----------------
// yT[j,cc] = sum_i sT[j][i] * dF[cc][i];  out[b][cc][j] = x[b][cc][j] + yT[j,cc]
// Double-buffered LDS staging via global_load_async_to_lds_b128 (ASYNCcnt).
// grid: (N/128 [j], C/128=4 [cc], B), block 128 (2x2 waves)
constexpr int LPAD = 40;                       // 32 + 8 halves row padding
constexpr unsigned LBUF_BYTES = 128u * LPAD * 2u;   // one buffer = 10240 B

__global__ __launch_bounds__(128)
void out_gemm(const _Float16* __restrict__ sT, const _Float16* __restrict__ dF,
              const float* __restrict__ x, float* __restrict__ out) {
    __shared__ _Float16 lsA[2][128 * LPAD];    // sT slice  [row j][k 0..31]
    __shared__ _Float16 lsB[2][128 * LPAD];    // d  slice  [col cc][k 0..31]

    int tid  = threadIdx.x;                    // 0..127
    int lane = tid & 31;
    int wave = tid >> 5;
    int wm = wave >> 1, wn = wave & 1;
    int batch  = blockIdx.z;
    int blockM = blockIdx.x * 128;             // j base
    int blockN = blockIdx.y * 128;             // cc base

    // per-thread staging sources: one row of A, one column of B (32 halves)
    const _Float16* gA = sT + (size_t)batch * NPIX * NPIX + (size_t)(blockM + tid) * NPIX;
    const _Float16* gB = dF + (size_t)batch * C_ * NPIX   + (size_t)(blockN + tid) * NPIX;
    unsigned laOff = (unsigned)(uintptr_t)(&lsA[0][tid * LPAD]);  // low 32b == LDS offset
    unsigned lbOff = (unsigned)(uintptr_t)(&lsB[0][tid * LPAD]);

    auto issue = [&](int buf, int k0) {
        unsigned la = laOff + (unsigned)buf * LBUF_BYTES;
        unsigned lb = lbOff + (unsigned)buf * LBUF_BYTES;
        const _Float16* pa = gA + k0;
        const _Float16* pb = gB + k0;
#pragma unroll
        for (int c = 0; c < 4; ++c) {          // 4 x 16B per row/col
            async_ld16(la + c * 16u, pa + c * 8);
            async_ld16(lb + c * 16u, pb + c * 8);
        }
    };

    v8f acc[4][4];
#pragma unroll
    for (int i = 0; i < 4; ++i)
#pragma unroll
        for (int j = 0; j < 4; ++j) acc[i][j] = (v8f)0.f;

    issue(0, 0);                               // prologue: fill buffer 0
    for (int it = 0; it < NPIX / 32; ++it) {
        int  cur  = it & 1;
        bool more = (it + 1) < NPIX / 32;
        if (more) issue(cur ^ 1, 32 * (it + 1));     // prefetch next K slice
        // wait for current buffer's 8 per-wave async loads (in-order retire)
        if (more) asm volatile("s_wait_asynccnt 0x8" ::: "memory");
        else      asm volatile("s_wait_asynccnt 0x0" ::: "memory");
        __syncthreads();                       // all waves' writes visible

        const _Float16* baseA = &lsA[cur][wm * 64 * LPAD];
        const _Float16* baseB = &lsB[cur][wn * 64 * LPAD];
        v16h af[4], bf[4];
#pragma unroll
        for (int i = 0; i < 4; ++i) af[i] = load_frag_a(baseA + (16 * i) * LPAD, LPAD, lane);
#pragma unroll
        for (int j = 0; j < 4; ++j) bf[j] = load_frag_b(baseB + (16 * j) * LPAD, LPAD, lane);
#pragma unroll
        for (int i = 0; i < 4; ++i)
#pragma unroll
            for (int j = 0; j < 4; ++j) acc[i][j] = wmma_f16(af[i], bf[j], acc[i][j]);

        __syncthreads();                       // reads done before buf reuse
    }

    int lg = lane >> 4, ln = lane & 15;
#pragma unroll
    for (int ti = 0; ti < 4; ++ti) {
        int jbase = blockM + wm * 64 + 16 * ti + 8 * lg;   // 8 contiguous j
#pragma unroll
        for (int tj = 0; tj < 4; ++tj) {
            int cc = blockN + wn * 64 + 16 * tj + ln;
            size_t off = ((size_t)batch * C_ + cc) * NPIX + jbase;
#pragma unroll
            for (int r = 0; r < 8; ++r)
                out[off + r] = x[off + r] + acc[ti][tj][r];
        }
    }
}

// ---------------------------------------------------------------------------
extern "C" void kernel_launch(void* const* d_in, const int* in_sizes, int n_in,
                              void* d_out, int out_size, void* d_ws, size_t ws_size,
                              hipStream_t stream) {
    (void)in_sizes; (void)n_in; (void)out_size; (void)ws_size;

    const float* x  = (const float*)d_in[0];
    const float* Wb = (const float*)d_in[1];
    const float* bb = (const float*)d_in[2];
    const float* Wc = (const float*)d_in[3];
    const float* bc = (const float*)d_in[4];
    const float* Wd = (const float*)d_in[5];
    const float* bd = (const float*)d_in[6];
    float* out = (float*)d_out;

    char* ws = (char*)d_ws;
    _Float16* xT      = (_Float16*)(ws + OFF_XT);
    _Float16* Wall    = (_Float16*)(ws + OFF_WALL);
    float*    biasAll = (float*)   (ws + OFF_BIAS);
    _Float16* bT      = (_Float16*)(ws + OFF_BT);
    _Float16* cT      = (_Float16*)(ws + OFF_CT);
    _Float16* dF      = (_Float16*)(ws + OFF_DF);
    _Float16* sT      = (_Float16*)(ws + OFF_ST);

    // 1. pack weights (640*512 elements)
    cast_weights<<<(MALL * C_) / 256, 256, 0, stream>>>(Wb, bb, Wc, bc, Wd, bd, Wall, biasAll);

    // 2. x -> xT f16
    cast_transpose_x<<<dim3(NPIX / 32, C_ / 32, B_), dim3(32, 8), 0, stream>>>(x, xT);

    // 3. fused projections
    proj_gemm<<<dim3(NPIX / 128, MALL / 128, B_), 128, 0, stream>>>(Wall, biasAll, xT, bT, cT, dF);

    // 4. logits (stored transposed, f16)
    logits_gemm<<<dim3(NPIX / 128, NPIX / 128, B_), 128, 0, stream>>>(bT, cT, sT);

    // 5. softmax over j, in place
    softmax_inplace<<<dim3(NPIX / 256, B_), 256, 0, stream>>>(sT);

    // 6. final GEMM + residual (async LDS double-buffer pipeline)
    out_gemm<<<dim3(NPIX / 128, C_ / 128, B_), 128, 0, stream>>>(sT, dF, x, out);
}
